// RelPositionMultiHeadAttention_27109833572473
// MI455X (gfx1250) — matrix operand order repaired and checked by reading, not compiled, not run on hardware
//
#include <hip/hip_runtime.h>

// ---------------- problem constants ----------------
constexpr int kB  = 8;
constexpr int kT  = 1024;
constexpr int kF  = 512;
constexpr int kH  = 8;
constexpr int kDK = 64;
constexpr int kP  = 2047;     // 2*T - 1
constexpr int kPpad = 2048;

typedef __attribute__((ext_vector_type(16))) _Float16     v16h;
typedef __attribute__((ext_vector_type(8)))  _Float16     v8h;
typedef __attribute__((ext_vector_type(8)))  float        v8f;
typedef __attribute__((ext_vector_type(4)))  unsigned int v4u;
typedef __attribute__((ext_vector_type(4)))  int          v4i;
typedef __attribute__((ext_vector_type(8)))  int          v8i;

#if defined(__has_builtin)
#if __has_builtin(__builtin_amdgcn_tensor_load_to_lds)
#define HAVE_TDM 1
#endif
#endif

__device__ __forceinline__ v8f wmma16(v16h a, v16h b, v8f c) {
  // D = A(16x32 f16) * B(32x16 f16) + C(16x16 f32)
  return __builtin_amdgcn_wmma_f32_16x16x32_f16(false, a, false, b, (short)0, c,
                                                false, false);
}

// 16 contiguous halves -> v16h (two b128 loads)
__device__ __forceinline__ v16h load16h(const _Float16* p) {
  union { v16h h; v4u u[2]; } r;
  r.u[0] = *(const v4u*)(p);
  r.u[1] = *(const v4u*)(p + 8);
  return r.h;
}
// A-fragment pattern: 8 halves at p, 8 halves at p+16 (two b128 loads)
__device__ __forceinline__ v16h load16h_split(const _Float16* p) {
  union { v16h h; v4u u[2]; } r;
  r.u[0] = *(const v4u*)(p);
  r.u[1] = *(const v4u*)(p + 16);
  return r.h;
}

#ifdef HAVE_TDM
// 1-line TDM copy: `elems` f16 elements from global -> LDS (D# per ISA ch.8)
__device__ __forceinline__ void tdm_load_1d(unsigned ldsAddr, const void* gptr,
                                            unsigned elems) {
  const unsigned long long ga = (unsigned long long)gptr;
  v4u g0;
  g0[0] = 1u;                                            // count=1, user D#
  g0[1] = ldsAddr;                                       // lds_addr (bytes)
  g0[2] = (unsigned)(ga & 0xFFFFFFFFull);                // global_addr[31:0]
  g0[3] = (unsigned)((ga >> 32) & 0x01FFFFFFu) | (2u << 30);  // addr[56:32] | type=2
  v8i g1;
  g1[0] = (int)(1u << 16);                               // data_size = 2 bytes
  g1[1] = (int)((elems & 0xFFFFu) << 16);                // tensor_dim0[15:0]
  g1[2] = (int)((elems >> 16) | (1u << 16));             // tensor_dim0[31:16] | tensor_dim1=1
  g1[3] = (int)((elems & 0xFFFFu) << 16);                // tile_dim0
  g1[4] = 1;                                             // tile_dim1 = 1
  g1[5] = (int)elems;                                    // tensor_dim0_stride
  g1[6] = 0;
  g1[7] = 0;
  v4i gz = {0, 0, 0, 0};
#if __clang_major__ >= 23
  v8i gz8 = {0, 0, 0, 0, 0, 0, 0, 0};
  __builtin_amdgcn_tensor_load_to_lds(g0, g1, gz, gz, gz8, 0);
#else
  __builtin_amdgcn_tensor_load_to_lds(g0, g1, gz, gz, 0);
#endif
}
#endif

// =====================================================================
// One-pass f32 -> f16 weight conversion (amortizes cvt across all tiles)
// =====================================================================
__global__ __launch_bounds__(256)
void cvtw_kernel(const float* __restrict__ src, _Float16* __restrict__ dst, int n)
{
  const int i = (blockIdx.x * 256 + threadIdx.x) * 8;
  if (i >= n) return;
  float4 x0 = *(const float4*)(src + i);
  float4 x1 = *(const float4*)(src + i + 4);
  v8h o;
  o[0]=(_Float16)x0.x; o[1]=(_Float16)x0.y; o[2]=(_Float16)x0.z; o[3]=(_Float16)x0.w;
  o[4]=(_Float16)x1.x; o[5]=(_Float16)x1.y; o[6]=(_Float16)x1.z; o[7]=(_Float16)x1.w;
  *(v8h*)(dst + i) = o;
}

// =====================================================================
// Projection GEMM: Y = X @ W^T (+bias), X:(M,512) f32, Wh:(512,512) f16.
// Wave computes a 16x64 tile (4 WMMAs per K-chunk, A reused).
// mode 0: Q  -> out0=q+pos_bias_u, out1=q+pos_bias_v, layout [bh][t][dk]
// mode 1: K  -> out0 layout [bh][t][dk]
// mode 2: V  -> out0 layout [bh][dk][t]   (transposed for PV B-operand)
// mode 3: P  -> out0 layout [h][j][dk], rows padded to 2048 (row 2047 = 0)
// =====================================================================
__global__ __launch_bounds__(128)
void proj_kernel(const float* __restrict__ X, const _Float16* __restrict__ Wh,
                 const float* __restrict__ bias,
                 const float* __restrict__ pbu, const float* __restrict__ pbv,
                 _Float16* __restrict__ out0, _Float16* __restrict__ out1,
                 int Mvalid, int mode)
{
  const int lane  = threadIdx.x & 31;
  const int wave  = threadIdx.x >> 5;
  const int tileM = blockIdx.x;
  const int tileN = blockIdx.y * 4 + wave;   // 64-col group; == head for QKV
  const int half  = lane >> 4;
  const int l15   = lane & 15;

  const int  rowA    = tileM * 16 + l15;
  const bool rowA_ok = rowA < Mvalid;
  const int   rowL = rowA_ok ? rowA : 0;     // clamp + zero-mul (no predication)
  const float zm   = rowA_ok ? 1.f : 0.f;

  const float* Xrow = X + (long)rowL * kF;
  const int colTile = tileN * 64;
  const _Float16* Wrow[4];
  #pragma unroll
  for (int n = 0; n < 4; ++n)
    Wrow[n] = Wh + (long)(colTile + n * 16 + l15) * kF;

  v8f acc[4] = {v8f{}, v8f{}, v8f{}, v8f{}};
  for (int kc = 0; kc < kF; kc += 32) {
    // A fragment: two runs of 8 at kc+half*8 and +16 (4 x b128 + cvt)
    const int abase = kc + half * 8;
    float4 a0 = *(const float4*)(Xrow + abase);
    float4 a1 = *(const float4*)(Xrow + abase + 4);
    float4 a2 = *(const float4*)(Xrow + abase + 16);
    float4 a3 = *(const float4*)(Xrow + abase + 20);
    v16h a;
    a[0]=(_Float16)(a0.x*zm); a[1]=(_Float16)(a0.y*zm);
    a[2]=(_Float16)(a0.z*zm); a[3]=(_Float16)(a0.w*zm);
    a[4]=(_Float16)(a1.x*zm); a[5]=(_Float16)(a1.y*zm);
    a[6]=(_Float16)(a1.z*zm); a[7]=(_Float16)(a1.w*zm);
    a[8]=(_Float16)(a2.x*zm); a[9]=(_Float16)(a2.y*zm);
    a[10]=(_Float16)(a2.z*zm); a[11]=(_Float16)(a2.w*zm);
    a[12]=(_Float16)(a3.x*zm); a[13]=(_Float16)(a3.y*zm);
    a[14]=(_Float16)(a3.z*zm); a[15]=(_Float16)(a3.w*zm);

    // Load all four f16 B fragments first (8 independent b128 loads),
    // then issue the four WMMAs.
    const int bbase = kc + half * 8;
    v16h bf[4];
    #pragma unroll
    for (int n = 0; n < 4; ++n) bf[n] = load16h_split(Wrow[n] + bbase);
    #pragma unroll
    for (int n = 0; n < 4; ++n) acc[n] = wmma16(a, bf[n], acc[n]);
  }

  // Epilogue: per-column values hoisted out of the row loop.
  #pragma unroll
  for (int n = 0; n < 4; ++n) {
    const int col = colTile + n * 16 + l15;
    const int h = col >> 6;          // uniform per wave for QKV
    const int d = col & 63;
    const float bval = bias ? bias[col] : 0.f;
    float uval = 0.f, vval = 0.f;
    if (mode == 0) { uval = pbu[h * kDK + d]; vval = pbv[h * kDK + d]; }
    #pragma unroll
    for (int r = 0; r < 8; ++r) {
      const int row = tileM * 16 + r + half * 8;
      const float val = acc[n][r] + bval;
      if (mode == 3) {
        out0[((long)h * kPpad + row) * kDK + d] = (_Float16)val;
      } else {
        const int bb = row >> 10;
        const int t  = row & (kT - 1);
        if (mode == 0) {
          const long off = (((long)(bb * kH + h)) * kT + t) * kDK + d;
          out0[off] = (_Float16)(val + uval);
          out1[off] = (_Float16)(val + vval);
        } else if (mode == 1) {
          out0[(((long)(bb * kH + h)) * kT + t) * kDK + d] = (_Float16)val;
        } else {
          out0[(((long)(bb * kH + h)) * kDK + d) * kT + t] = (_Float16)val;
        }
      }
    }
  }
}

// =====================================================================
// Fused rel-pos flash attention (one wave per (b,h,16-row q tile)).
// K tile + mask tile staged via async-to-LDS; p band via TDM (fallback async).
// =====================================================================
__global__ __launch_bounds__(32)
void attn_kernel(const _Float16* __restrict__ qu, const _Float16* __restrict__ qv,
                 const _Float16* __restrict__ kb, const _Float16* __restrict__ vt,
                 const _Float16* __restrict__ pb, const unsigned char* __restrict__ mask,
                 _Float16* __restrict__ xout)
{
  const int lane = threadIdx.x & 31;
  const int half = lane >> 4;
  const int l15  = lane & 15;

  const int wg = blockIdx.x;          // (b*H + h) * (T/16) + qtile
  const int qt = wg & 63;             // T/16 == 64
  const int bh = wg >> 6;
  const int b  = bh >> 3;
  const int h  = bh & 7;
  const int q0 = qt * 16;

  __shared__ float         lds_g[16 * 48];
  __shared__ _Float16      lds_p[16 * 32];
  __shared__ _Float16      lds_k[32 * 64];     // staged K tile (4KB)
  __shared__ _Float16      lds_band[48 * 64];  // staged p band (6KB)
  __shared__ unsigned char lds_m[16 * 32];     // staged mask tile (512B)

  const _Float16* quRow = qu + (((long)bh) * kT + q0 + l15) * kDK;
  const _Float16* qvRow = qv + (((long)bh) * kT + q0 + l15) * kDK;
  v16h au[2], av[2];
  #pragma unroll
  for (int c = 0; c < 2; ++c) {
    au[c] = load16h_split(quRow + c * 32 + half * 8);
    av[c] = load16h_split(qvRow + c * 32 + half * 8);
  }

  v8f O[4] = {v8f{}, v8f{}, v8f{}, v8f{}};   // 16 x 64 accumulator
  float rowmax[8], rowsum[8];
  #pragma unroll
  for (int r = 0; r < 8; ++r) { rowmax[r] = -1e30f; rowsum[r] = 0.f; }

  const _Float16* kbh = kb + ((long)bh) * kT * kDK;       // [t][dk]
  const _Float16* vth = vt + ((long)bh) * kDK * kT;       // [dk][t]
  const _Float16* pbh = pb + ((long)h) * kPpad * kDK;     // [j][dk]
  const unsigned char* mb = mask + ((long)b) * kT * kT;

  const unsigned ldsKBase = (unsigned)(unsigned long long)(&lds_k[0]);
  const unsigned ldsBBase = (unsigned)(unsigned long long)(&lds_band[0]);
  const unsigned ldsMBase = (unsigned)(unsigned long long)(&lds_m[0]);

  for (int k0 = 0; k0 < kT; k0 += 32) {
    const int jbase = k0 - q0 + (kT - 16);   // k0 - q0 + 1008, always >= 0

    // ---- async DMA: K tile (32 x 64 halves = 4KB) into LDS
    {
      const char* gk = (const char*)(kbh + (long)k0 * kDK);
      #pragma unroll
      for (int i = 0; i < 8; ++i) {
        const unsigned loff = ldsKBase + (unsigned)(lane + i * 32) * 16u;
        const char*    ga   = gk + (lane + i * 32) * 16;
        asm volatile("global_load_async_to_lds_b128 %0, %1, off"
                     :: "v"(loff), "v"(ga) : "memory");
      }
    }
    // ---- async DMA: mask tile (16 rows x 32 bytes) in ONE instruction
    {
      const int mrow = lane >> 1;          // 0..15
      const int mcol = (lane & 1) * 16;    // 0 or 16
      const char*    ga   = (const char*)(mb + (long)(q0 + mrow) * kT + k0 + mcol);
      const unsigned loff = ldsMBase + (unsigned)(mrow * 32 + mcol);
      asm volatile("global_load_async_to_lds_b128 %0, %1, off"
                   :: "v"(loff), "v"(ga) : "memory");
    }
    // ---- TDM: p band (48 rows x 64 halves, contiguous 6KB) into LDS
#ifdef HAVE_TDM
    tdm_load_1d(ldsBBase, pbh + (long)jbase * kDK, 48 * 64);
#else
    {
      const char* gp = (const char*)(pbh + (long)jbase * kDK);
      #pragma unroll
      for (int i = 0; i < 12; ++i) {
        const unsigned loff = ldsBBase + (unsigned)(lane + i * 32) * 16u;
        const char*    ga   = gp + (lane + i * 32) * 16;
        asm volatile("global_load_async_to_lds_b128 %0, %1, off"
                     :: "v"(loff), "v"(ga) : "memory");
      }
    }
#endif
    if (k0 + 32 < kT)
      __builtin_prefetch(kbh + (long)(k0 + 32) * kDK, 0, 1);

    // ---- wait p band, compute G = qv . p over the band
#ifdef HAVE_TDM
    __builtin_amdgcn_s_wait_tensorcnt((short)0);
#else
    asm volatile("s_wait_asynccnt 0x0" ::: "memory");
#endif
    __syncthreads();
    #pragma unroll
    for (int s = 0; s < 3; ++s) {
      v8f G = {};
      const _Float16* prow = &lds_band[(s * 16 + l15) * 64];
      #pragma unroll
      for (int c = 0; c < 2; ++c)
        G = wmma16(av[c], load16h_split(prow + c * 32 + half * 8), G);
      #pragma unroll
      for (int r = 0; r < 8; ++r)
        lds_g[(r + half * 8) * 48 + s * 16 + l15] = G[r];
    }

    // ---- wait K + mask tiles, AC from LDS
    asm volatile("s_wait_asynccnt 0x0" ::: "memory");
    __syncthreads();
    v8f S[2] = {v8f{}, v8f{}};
    #pragma unroll
    for (int st = 0; st < 2; ++st) {
      const _Float16* krow = &lds_k[(st * 16 + l15) * 64];
      #pragma unroll
      for (int c = 0; c < 2; ++c)
        S[st] = wmma16(au[c], load16h_split(krow + c * 32 + half * 8), S[st]);
    }

    // ---- combine AC+BD, mask, online softmax
    float sc[2][8]; bool mk[2][8];
    float newmax[8];
    #pragma unroll
    for (int r = 0; r < 8; ++r) newmax[r] = rowmax[r];
    #pragma unroll
    for (int st = 0; st < 2; ++st) {
      const int kk = st * 16 + l15;
      #pragma unroll
      for (int r = 0; r < 8; ++r) {
        const int row = r + half * 8;
        const float bd = lds_g[row * 48 + (kk - row + 15)];   // rel_shift gather
        float s = (S[st][r] + bd) * 0.125f;                   // 1/sqrt(64)
        const bool m = lds_m[row * 32 + kk] != 0;
        if (m) s = -10000.f;
        sc[st][r] = s; mk[st][r] = m;
        newmax[r] = fmaxf(newmax[r], s);
      }
    }
    #pragma unroll
    for (int r = 0; r < 8; ++r) {
      float v = newmax[r];
      #pragma unroll
      for (int m = 1; m < 16; m <<= 1) v = fmaxf(v, __shfl_xor(v, m, 32));
      newmax[r] = v;
    }
    float rscale[8];
    #pragma unroll
    for (int r = 0; r < 8; ++r) {
      rscale[r] = __expf(rowmax[r] - newmax[r]);
      rowmax[r] = newmax[r];
    }
    float e[2][8], psum[8];
    #pragma unroll
    for (int r = 0; r < 8; ++r) psum[r] = 0.f;
    #pragma unroll
    for (int st = 0; st < 2; ++st)
      #pragma unroll
      for (int r = 0; r < 8; ++r) {
        const float ev = mk[st][r] ? 0.f : __expf(sc[st][r] - rowmax[r]);
        e[st][r] = ev;
        psum[r] += ev;
      }
    #pragma unroll
    for (int r = 0; r < 8; ++r) {
      float v = psum[r];
      #pragma unroll
      for (int m = 1; m < 16; m <<= 1) v += __shfl_xor(v, m, 32);
      rowsum[r] = rowsum[r] * rscale[r] + v;
    }
    #pragma unroll
    for (int dn = 0; dn < 4; ++dn)
      #pragma unroll
      for (int r = 0; r < 8; ++r)
        O[dn][r] *= rscale[r];

    // ---- repack P tile (C layout) into A-fragment layout via LDS
    #pragma unroll
    for (int st = 0; st < 2; ++st)
      #pragma unroll
      for (int r = 0; r < 8; ++r)
        lds_p[(r + half * 8) * 32 + st * 16 + l15] = (_Float16)e[st][r];
    __syncthreads();
    const v16h pa = load16h_split(&lds_p[l15 * 32] + half * 8);

    // ---- P @ V : O[:, dn*16..] += P(16x32) * V(32x16)
    #pragma unroll
    for (int dn = 0; dn < 4; ++dn) {
      const _Float16* vrow = vth + (long)(dn * 16 + l15) * kT + k0 + half * 8;
      O[dn] = wmma16(pa, load16h_split(vrow), O[dn]);
    }
    __syncthreads();   // before next iteration rewrites LDS buffers
  }

  // ---- normalize and store x[b][t][h][dk] as f16
  #pragma unroll
  for (int r = 0; r < 8; ++r) {
    const int row = r + half * 8;
    const float inv = rowsum[r] > 0.f ? 1.f / rowsum[r] : 0.f;
    #pragma unroll
    for (int dn = 0; dn < 4; ++dn) {
      const int d = dn * 16 + l15;
      const long off = (((long)b * kT + q0 + row) * kH + h) * kDK + d;
      xout[off] = (_Float16)(O[dn][r] * inv);
    }
  }
}

// =====================================================================
// Output projection: Y = Xh(f16) @ Woh^T + bo -> f32 (16x64 tile per wave)
// =====================================================================
__global__ __launch_bounds__(128)
void outproj_kernel(const _Float16* __restrict__ Xh, const _Float16* __restrict__ Wh,
                    const float* __restrict__ bias, float* __restrict__ Y)
{
  const int lane  = threadIdx.x & 31;
  const int wave  = threadIdx.x >> 5;
  const int tileM = blockIdx.x;
  const int tileN = blockIdx.y * 4 + wave;
  const int half  = lane >> 4;
  const int l15   = lane & 15;
  const int rowA  = tileM * 16 + l15;
  const int colTile = tileN * 64;

  const _Float16* Xrow = Xh + (long)rowA * kF;
  const _Float16* Wrow[4];
  #pragma unroll
  for (int n = 0; n < 4; ++n)
    Wrow[n] = Wh + (long)(colTile + n * 16 + l15) * kF;

  v8f acc[4] = {v8f{}, v8f{}, v8f{}, v8f{}};
  for (int kc = 0; kc < kF; kc += 32) {
    v16h a = load16h_split(Xrow + kc + half * 8);
    const int bbase = kc + half * 8;
    v16h bf[4];
    #pragma unroll
    for (int n = 0; n < 4; ++n) bf[n] = load16h_split(Wrow[n] + bbase);
    #pragma unroll
    for (int n = 0; n < 4; ++n) acc[n] = wmma16(a, bf[n], acc[n]);
  }
  #pragma unroll
  for (int n = 0; n < 4; ++n) {
    const int col = colTile + n * 16 + l15;
    const float bval = bias[col];
    #pragma unroll
    for (int r = 0; r < 8; ++r) {
      const int row = tileM * 16 + r + half * 8;
      Y[(long)row * kF + col] = acc[n][r] + bval;
    }
  }
}

// =====================================================================
extern "C" void kernel_launch(void* const* d_in, const int* in_sizes, int n_in,
                              void* d_out, int out_size, void* d_ws, size_t ws_size,
                              hipStream_t stream)
{
  const float* query   = (const float*)d_in[0];
  const float* key     = (const float*)d_in[1];
  const float* value   = (const float*)d_in[2];
  const unsigned char* mask = (const unsigned char*)d_in[3];
  const float* pos_emb = (const float*)d_in[4];
  const float* Wq  = (const float*)d_in[5];
  const float* bq  = (const float*)d_in[6];
  const float* Wk  = (const float*)d_in[7];
  const float* bk  = (const float*)d_in[8];
  const float* Wv  = (const float*)d_in[9];
  const float* bv  = (const float*)d_in[10];
  const float* Wpos= (const float*)d_in[11];
  const float* Wo  = (const float*)d_in[12];
  const float* bo  = (const float*)d_in[13];
  const float* pbu = (const float*)d_in[14];
  const float* pbv = (const float*)d_in[15];

  // workspace carve (f16)
  char* ws = (char*)d_ws;
  const size_t qkvElems = (size_t)kB * kH * kT * kDK;       // 4,194,304
  const size_t wElems   = (size_t)kF * kF;                  // 262,144
  _Float16* q_u  = (_Float16*)ws; ws += qkvElems * 2;
  _Float16* q_v  = (_Float16*)ws; ws += qkvElems * 2;
  _Float16* kbuf = (_Float16*)ws; ws += qkvElems * 2;
  _Float16* vtb  = (_Float16*)ws; ws += qkvElems * 2;
  _Float16* pbuf = (_Float16*)ws; ws += (size_t)kH * kPpad * kDK * 2;
  _Float16* xbuf = (_Float16*)ws; ws += (size_t)kB * kT * kF * 2;
  _Float16* Wqh  = (_Float16*)ws; ws += wElems * 2;
  _Float16* Wkh  = (_Float16*)ws; ws += wElems * 2;
  _Float16* Wvh  = (_Float16*)ws; ws += wElems * 2;
  _Float16* Wph  = (_Float16*)ws; ws += wElems * 2;
  _Float16* Woh  = (_Float16*)ws; ws += wElems * 2;

  // ---- one-pass weight conversion to f16
  const int nW = (int)wElems;
  const dim3 cgrid((nW / 8 + 255) / 256), cblk(256);
  cvtw_kernel<<<cgrid, cblk, 0, stream>>>(Wq,   Wqh, nW);
  cvtw_kernel<<<cgrid, cblk, 0, stream>>>(Wk,   Wkh, nW);
  cvtw_kernel<<<cgrid, cblk, 0, stream>>>(Wv,   Wvh, nW);
  cvtw_kernel<<<cgrid, cblk, 0, stream>>>(Wpos, Wph, nW);
  cvtw_kernel<<<cgrid, cblk, 0, stream>>>(Wo,   Woh, nW);

  const dim3 blk(128);
  const dim3 gQKV(kB * kT / 16, kF / 256);     // (512, 2)
  proj_kernel<<<gQKV, blk, 0, stream>>>(query, Wqh, bq, pbu, pbv, q_u, q_v,
                                        kB * kT, 0);
  proj_kernel<<<gQKV, blk, 0, stream>>>(key, Wkh, bk, nullptr, nullptr, kbuf,
                                        nullptr, kB * kT, 1);
  proj_kernel<<<gQKV, blk, 0, stream>>>(value, Wvh, bv, nullptr, nullptr, vtb,
                                        nullptr, kB * kT, 2);
  const dim3 gP(kPpad / 16, kF / 256);         // (128, 2)
  proj_kernel<<<gP, blk, 0, stream>>>(pos_emb, Wph, nullptr, nullptr, nullptr,
                                      pbuf, nullptr, kP, 3);

  attn_kernel<<<dim3(kB * kH * (kT / 16)), dim3(32), 0, stream>>>(
      q_u, q_v, kbuf, vtb, pbuf, mask, xbuf);

  outproj_kernel<<<dim3(kB * kT / 16, kF / 256), blk, 0, stream>>>(
      xbuf, Woh, bo, (float*)d_out);
}
